// VisionTransformer_44409961841143
// MI455X (gfx1250) — matrix operand order repaired and checked by reading, not compiled, not run on hardware
//
#include <hip/hip_runtime.h>

// ============================================================================
// ViT-B/16 forward for MI455X (gfx1250, wave32, WMMA bf16 -> f32 accumulate)
//
// - All matmuls via v_wmma_f32_16x16x32_bf16 (fp32 accum).
// - Weights fp32 are transposed+converted once per use into bf16 [N,K] so the
//   GEMM stages both operands as contiguous 16B chunks.
// - Global->LDS staging uses GLOBAL_LOAD_ASYNC_TO_LDS_B128 when the builtin
//   exists (ASYNCcnt + s_wait_asynccnt), else a synchronous fallback.
// ============================================================================

typedef __bf16 bf16;
typedef __bf16 v8bf  __attribute__((ext_vector_type(8)));
typedef __bf16 v16bf __attribute__((ext_vector_type(16)));
typedef float  v8f   __attribute__((ext_vector_type(8)));
typedef int    v4i   __attribute__((vector_size(16)));
typedef unsigned long long ull;

#define B_   32
#define D_   768
#define F_   3072
#define L_   12
#define MTOK 6304     // 32 * 197
#define MPAT 6272     // 32 * 196

#if __has_builtin(__builtin_amdgcn_global_load_async_to_lds_b128)
#define HAVE_ASYNC_LDS 1
#else
#define HAVE_ASYNC_LDS 0
#endif

static __device__ __forceinline__ bf16 f2bf(float f) {
  unsigned u = __builtin_bit_cast(unsigned, f);
  unsigned r = (u + 0x7FFFu + ((u >> 16) & 1u)) >> 16;
  unsigned short h = (unsigned short)r;
  return __builtin_bit_cast(bf16, h);
}

// 16-byte global -> LDS copy (async on CDNA5 when available)
static __device__ __forceinline__ void cp16(const bf16* g, bf16* l) {
#if HAVE_ASYNC_LDS
  __builtin_amdgcn_global_load_async_to_lds_b128((v4i*)(void*)g, (v4i*)(void*)l,
                                                 0, 0);
#else
  *(uint4*)l = *(const uint4*)g;
#endif
}

static __device__ __forceinline__ void wait_async_lds() {
#if HAVE_ASYNC_LDS
#if __has_builtin(__builtin_amdgcn_s_wait_asynccnt)
  __builtin_amdgcn_s_wait_asynccnt(0);
#else
  asm volatile("s_wait_asynccnt 0" ::: "memory");
#endif
#endif
}

// ----------------------------------------------------------------------------
// Weight transpose+convert: in fp32 (element (k,n)) -> out bf16 [N,K] row-major
// MODE 0: in[k*N + n]   MODE 1 (QKV head-blocked): in[((n>>6)*768 + k)*64 + (n&63)]
// ----------------------------------------------------------------------------
template <int MODE>
__global__ void __launch_bounds__(256) k_wT(const float* __restrict__ in,
                                            bf16* __restrict__ out, int K, int N) {
  __shared__ float t[32][33];
  int tx = threadIdx.x & 31, ty = threadIdx.x >> 5;  // 32 x 8
  int n0 = blockIdx.x * 32, k0 = blockIdx.y * 32;
#pragma unroll
  for (int r = 0; r < 4; ++r) {
    int k = k0 + ty + r * 8, n = n0 + tx;
    float v = 0.f;
    if (k < K && n < N) {
      if (MODE == 0) v = in[(size_t)k * N + n];
      else           v = in[((size_t)(n >> 6) * 768 + k) * 64 + (n & 63)];
    }
    t[ty + r * 8][tx] = v;
  }
  __syncthreads();
#pragma unroll
  for (int r = 0; r < 4; ++r) {
    int n = n0 + ty + r * 8, k = k0 + tx;
    if (n < N && k < K) out[(size_t)n * K + k] = f2bf(t[tx][ty + r * 8]);
  }
}

// ----------------------------------------------------------------------------
// Tiled WMMA GEMM: C[M,N] = A[M,K] * Bt[N,K]^T  (both operands bf16)
// Block 256 threads (8 waves), tile 128x128, BK=64 (two wmma k-steps/stage).
// Wave w: rows (w>>1)*32, cols (w&1)*64  -> 2x4 fragments, 16 WMMAs/stage.
// EP (epilogue):
//   0: outF[z*obs + m*ldo + n] = acc*scale + bias[n]     (bias may be null)
//   1: outF[m*ldo + n] += acc + bias[n]                  (residual, in-place)
//   2: outB[m*ldo + n] = bf16(gelu_exact(acc + bias[n]))
//   3: Q/K scatter:  outB[((b*12+h)*197+s)*64 + e] = bf16(acc + bias[n])
//   4: V^T scatter:  outB[((b*12+h)*64+e)*208 + s] = bf16(acc + bias[n])
//   5: patch embed:  outF[(b*197+s+1)*768 + n] = acc + bias[n] + pos[s*768+n]
//   6: attn out:     outB[(b*197+m)*768 + h*64 + n] = bf16(acc)  (b,h from z)
// ----------------------------------------------------------------------------
template <int EP>
__global__ void __launch_bounds__(256) gemm_k(
    const bf16* __restrict__ A, int lda, ull a_bs,
    const bf16* __restrict__ Bt, int ldb, ull b_bs,
    int M, int N, int K,
    float* __restrict__ outF, bf16* __restrict__ outB, int ldo, ull o_bs,
    const float* __restrict__ bias, const float* __restrict__ pos, float scale) {
  __shared__ bf16 As[128][72];  // 64 cols + 8 pad
  __shared__ bf16 Bs[128][72];

  const int tid = threadIdx.x;
  const int z = blockIdx.z;
  const int m0 = blockIdx.y * 128;
  const int n0 = blockIdx.x * 128;
  const bf16* Az = A + (size_t)z * a_bs;
  const bf16* Bz = Bt + (size_t)z * b_bs;

  const int w = tid >> 5, lane = tid & 31;
  const int wm = (w >> 1) * 32, wn = (w & 1) * 64;
  const int half = lane >> 4, lr = lane & 15;

  // staging coords: each thread stages one 32-elem half-row of A and of B
  const int srow = tid >> 1, scb = (tid & 1) * 32;
  const int agm = m0 + srow < M ? m0 + srow : M - 1;  // clamp: OOB rows only
  const int bgn = n0 + srow < N ? n0 + srow : N - 1;  // feed discarded outputs
  const bf16* arow = Az + (size_t)agm * lda + scb;
  const bf16* brow = Bz + (size_t)bgn * ldb + scb;

  v8f acc[2][4];
#pragma unroll
  for (int i = 0; i < 2; ++i)
#pragma unroll
    for (int j = 0; j < 4; ++j)
      acc[i][j] = (v8f){0.f, 0.f, 0.f, 0.f, 0.f, 0.f, 0.f, 0.f};

  for (int k0 = 0; k0 < K; k0 += 64) {
    __syncthreads();
    // ---- stage A & B tiles: 4 x 16B chunks each ----
#pragma unroll
    for (int c = 0; c < 4; ++c) {
      int gk = k0 + scb + c * 8;
      if (gk < K) {  // K-tail only (K%64 != 0 for attn V path)
        cp16(arow + k0 + c * 8, &As[srow][scb + c * 8]);
        cp16(brow + k0 + c * 8, &Bs[srow][scb + c * 8]);
      } else {
        *(uint4*)&As[srow][scb + c * 8] = (uint4){0u, 0u, 0u, 0u};
        *(uint4*)&Bs[srow][scb + c * 8] = (uint4){0u, 0u, 0u, 0u};
      }
    }
    if (k0 + 64 < K) {  // prefetch next K-tile (global_prefetch_b8)
      __builtin_prefetch(arow + k0 + 64, 0, 3);
      __builtin_prefetch(brow + k0 + 64, 0, 3);
    }
    wait_async_lds();
    __syncthreads();

    // ---- two 16x16x32 k-steps ----
#pragma unroll
    for (int ks = 0; ks < 2; ++ks) {
      const int kb = ks * 32;
      v16bf af[2], bfr[4];
#pragma unroll
      for (int i = 0; i < 2; ++i) {
        int r = wm + i * 16 + lr;
        v8bf lo = *(const v8bf*)&As[r][kb + half * 8];       // K = half*8..+7
        v8bf hi = *(const v8bf*)&As[r][kb + 16 + half * 8];  // K = 16+half*8..+7
        af[i] = __builtin_shufflevector(lo, hi, 0, 1, 2, 3, 4, 5, 6, 7, 8, 9,
                                        10, 11, 12, 13, 14, 15);
      }
#pragma unroll
      for (int j = 0; j < 4; ++j) {
        int c = wn + j * 16 + lr;
        v8bf lo = *(const v8bf*)&Bs[c][kb + half * 8];       // K = half*8..+7
        v8bf hi = *(const v8bf*)&Bs[c][kb + 16 + half * 8];  // K = 16+half*8..+7
        bfr[j] = __builtin_shufflevector(lo, hi, 0, 1, 2, 3, 4, 5, 6, 7, 8, 9,
                                         10, 11, 12, 13, 14, 15);
      }
#pragma unroll
      for (int i = 0; i < 2; ++i)
#pragma unroll
        for (int j = 0; j < 4; ++j)
          acc[i][j] = __builtin_amdgcn_wmma_f32_16x16x32_bf16(
              false, af[i], false, bfr[j], (short)0, acc[i][j], false, false);
    }
  }

  // ---- epilogue ----
#pragma unroll
  for (int i = 0; i < 2; ++i)
#pragma unroll
    for (int j = 0; j < 4; ++j)
#pragma unroll
      for (int r = 0; r < 8; ++r) {
        int gm = m0 + wm + i * 16 + half * 8 + r;
        int gn = n0 + wn + j * 16 + lr;
        if (gm >= M || gn >= N) continue;
        float a = acc[i][j][r];
        if (EP == 0) {
          float v = a * scale + (bias ? bias[gn] : 0.f);
          outF[(size_t)z * o_bs + (size_t)gm * ldo + gn] = v;
        } else if (EP == 1) {
          size_t idx = (size_t)gm * ldo + gn;
          outF[idx] = outF[idx] + a + bias[gn];
        } else if (EP == 2) {
          float v = a + bias[gn];
          v = 0.5f * v * (1.f + erff(v * 0.70710678118654752f));
          outB[(size_t)gm * ldo + gn] = f2bf(v);
        } else if (EP == 3) {
          int b = gm / 197, s = gm % 197, hh = gn >> 6, e = gn & 63;
          outB[(((size_t)(b * 12 + hh)) * 197 + s) * 64 + e] = f2bf(a + bias[gn]);
        } else if (EP == 4) {
          int b = gm / 197, s = gm % 197, hh = gn >> 6, e = gn & 63;
          outB[(((size_t)(b * 12 + hh)) * 64 + e) * 208 + s] = f2bf(a + bias[gn]);
        } else if (EP == 5) {
          int b = gm / 196, s = gm % 196;
          size_t row = (size_t)b * 197 + s + 1;
          outF[row * 768 + gn] = a + bias[gn] + pos[(size_t)s * 768 + gn];
        } else if (EP == 6) {
          int b = z / 12, hh = z % 12;
          outB[((size_t)(b * 197 + gm)) * 768 + hh * 64 + gn] = f2bf(a);
        }
      }
}

// ---- im2col: images [B,224,224,3] -> patches bf16 [B*196, 768] -------------
__global__ void k_im2col(const float* __restrict__ img, bf16* __restrict__ out) {
  int idx = blockIdx.x * 256 + threadIdx.x;
  if (idx >= MPAT * D_) return;
  int kcol = idx % 768, mp = idx / 768;
  int b = mp / 196, p = mp % 196;
  int py = p / 14, px = p % 14;
  int r = kcol / 48, rem = kcol % 48;
  int c = rem / 3, ch = rem % 3;
  float v = img[(((size_t)b * 224 + py * 16 + r) * 224 + (px * 16 + c)) * 3 + ch];
  out[idx] = f2bf(v);
}

// ---- cls token into x[:,0,:] ----------------------------------------------
__global__ void k_cls(const float* __restrict__ cls, float* __restrict__ x) {
  int idx = blockIdx.x * 256 + threadIdx.x;
  if (idx >= B_ * D_) return;
  int b = idx / 768, d = idx % 768;
  x[(size_t)b * 197 * 768 + d] = cls[d];
}

// ---- zero V^T padding columns (s = 197..207) -------------------------------
__global__ void k_zero_vpad(bf16* __restrict__ v) {
  int idx = blockIdx.x * 256 + threadIdx.x;
  if (idx >= 384 * 64 * 11) return;
  int zz = idx / (64 * 11), r = idx % (64 * 11);
  int e = r / 11, s = 197 + r % 11;
  v[((size_t)zz * 64 + e) * 208 + s] = f2bf(0.f);
}

// ---- LayerNorm: fp32 rows -> bf16 rows (one wave32 per row) ----------------
__global__ void k_ln(const float* __restrict__ x, long long row_stride,
                     const float* __restrict__ g, const float* __restrict__ b,
                     bf16* __restrict__ out, int nrows) {
  int w = threadIdx.x >> 5, lane = threadIdx.x & 31;
  int row = blockIdx.x * 8 + w;
  if (row >= nrows) return;
  const float* xp = x + (size_t)row * row_stride;
  float s = 0.f, s2 = 0.f, v[24];
#pragma unroll
  for (int i = 0; i < 24; ++i) {
    float t = xp[lane + i * 32];
    v[i] = t;
    s += t;
    s2 += t * t;
  }
#pragma unroll
  for (int off = 16; off > 0; off >>= 1) {
    s += __shfl_xor(s, off, 32);
    s2 += __shfl_xor(s2, off, 32);
  }
  float m = s * (1.f / 768.f);
  float var = s2 * (1.f / 768.f) - m * m;
  float rinv = rsqrtf(var + 1e-5f);
  bf16* op = out + (size_t)row * 768;
#pragma unroll
  for (int i = 0; i < 24; ++i) {
    int c = lane + i * 32;
    op[c] = f2bf((v[i] - m) * rinv * g[c] + b[c]);
  }
}

// ---- softmax over 197 (padded 208) fp32 -> bf16 (one wave32 per row) -------
__global__ void k_softmax(const float* __restrict__ S, bf16* __restrict__ att) {
  int w = threadIdx.x >> 5, lane = threadIdx.x & 31;
  int row = blockIdx.x * 8 + w;
  if (row >= 384 * 197) return;
  int z = row / 197, s = row % 197;
  const float* sp = S + (size_t)z * 208 * 208 + (size_t)s * 208;
  float vals[7], mx = -1e30f;
#pragma unroll
  for (int i = 0; i < 7; ++i) {
    int c = lane + i * 32;
    vals[i] = (c < 197) ? sp[c] : -1e30f;
    mx = fmaxf(mx, vals[i]);
  }
#pragma unroll
  for (int off = 16; off > 0; off >>= 1) mx = fmaxf(mx, __shfl_xor(mx, off, 32));
  float sum = 0.f;
#pragma unroll
  for (int i = 0; i < 7; ++i) {
    int c = lane + i * 32;
    float e = (c < 197) ? __expf(vals[i] - mx) : 0.f;
    vals[i] = e;
    sum += e;
  }
#pragma unroll
  for (int off = 16; off > 0; off >>= 1) sum += __shfl_xor(sum, off, 32);
  float inv = 1.f / sum;
  bf16* ap = att + (size_t)z * 208 * 208 + (size_t)s * 208;
#pragma unroll
  for (int i = 0; i < 7; ++i) {
    int c = lane + i * 32;
    if (c < 208) ap[c] = f2bf((c < 197) ? vals[i] * inv : 0.f);
  }
}

// ============================================================================
extern "C" void kernel_launch(void* const* d_in, const int* in_sizes, int n_in,
                              void* d_out, int out_size, void* d_ws,
                              size_t ws_size, hipStream_t stream) {
  (void)in_sizes; (void)n_in; (void)out_size; (void)ws_size;
  const float* images    = (const float*)d_in[0];
  const float* patch_W   = (const float*)d_in[1];
  const float* patch_b   = (const float*)d_in[2];
  const float* pos_emb   = (const float*)d_in[3];
  const float* cls       = (const float*)d_in[4];
  const float* ln1_g     = (const float*)d_in[5];
  const float* ln1_b     = (const float*)d_in[6];
  const float* Wq        = (const float*)d_in[7];
  const float* bq        = (const float*)d_in[8];
  const float* Wk        = (const float*)d_in[9];
  const float* bk        = (const float*)d_in[10];
  const float* Wv        = (const float*)d_in[11];
  const float* bv        = (const float*)d_in[12];
  const float* Wo        = (const float*)d_in[13];
  const float* bo        = (const float*)d_in[14];
  const float* ln2_g     = (const float*)d_in[15];
  const float* ln2_b     = (const float*)d_in[16];
  const float* W1        = (const float*)d_in[17];
  const float* b1        = (const float*)d_in[18];
  const float* W2        = (const float*)d_in[19];
  const float* b2        = (const float*)d_in[20];
  const float* head_g    = (const float*)d_in[21];
  const float* head_bn   = (const float*)d_in[22];
  const float* head_W    = (const float*)d_in[23];
  const float* head_bias = (const float*)d_in[24];

  // ---- workspace carve ----
  char* ws = (char*)d_ws;
  size_t off = 0;
  auto carve = [&](size_t bytes) -> void* {
    void* p = ws + off;
    off += (bytes + 255) & ~(size_t)255;
    return p;
  };
  float* x    = (float*)carve((size_t)MTOK * D_ * 4);
  bf16* xn    = (bf16*)carve((size_t)MTOK * D_ * 2);
  bf16* q     = (bf16*)carve((size_t)384 * 197 * 64 * 2);
  bf16* kb    = (bf16*)carve((size_t)384 * 197 * 64 * 2);
  bf16* vb    = (bf16*)carve((size_t)384 * 64 * 208 * 2);   // V^T [bh,64,208]
  float* S    = (float*)carve((size_t)384 * 208 * 208 * 4);
  bf16* att   = (bf16*)carve((size_t)384 * 208 * 208 * 2);
  bf16* ob    = (bf16*)carve((size_t)MTOK * D_ * 2);
  bf16* hmid  = (bf16*)carve((size_t)MTOK * F_ * 2);
  bf16* patch = (bf16*)carve((size_t)MPAT * D_ * 2);
  bf16* xcls  = (bf16*)carve((size_t)B_ * D_ * 2);
  // transposed bf16 weights (per-layer reuse)
  bf16* wt_p = (bf16*)carve((size_t)768 * 768 * 2);
  bf16* wt_q = (bf16*)carve((size_t)768 * 768 * 2);
  bf16* wt_k = (bf16*)carve((size_t)768 * 768 * 2);
  bf16* wt_v = (bf16*)carve((size_t)768 * 768 * 2);
  bf16* wt_o = (bf16*)carve((size_t)768 * 768 * 2);
  bf16* wt_1 = (bf16*)carve((size_t)3072 * 768 * 2);
  bf16* wt_2 = (bf16*)carve((size_t)768 * 3072 * 2);
  bf16* wt_h = (bf16*)carve((size_t)1000 * 768 * 2);

  const dim3 blk(256);

  // ---- preprocessing ----
  k_im2col<<<(MPAT * D_ + 255) / 256, blk, 0, stream>>>(images, patch);
  k_cls<<<(B_ * D_ + 255) / 256, blk, 0, stream>>>(cls, x);
  k_zero_vpad<<<(384 * 64 * 11 + 255) / 256, blk, 0, stream>>>(vb);

  k_wT<0><<<dim3(24, 24), blk, 0, stream>>>(patch_W, wt_p, 768, 768);
  gemm_k<5><<<dim3(6, 49, 1), blk, 0, stream>>>(
      patch, 768, 0, wt_p, 768, 0, MPAT, 768, 768, x, nullptr, 768, 0, patch_b,
      pos_emb, 1.f);

  const int lnBlocks = (MTOK + 7) / 8;
  for (int l = 0; l < L_; ++l) {
    const float* Wq_l = Wq + (size_t)l * 768 * 768;
    const float* Wk_l = Wk + (size_t)l * 768 * 768;
    const float* Wv_l = Wv + (size_t)l * 768 * 768;
    const float* Wo_l = Wo + (size_t)l * 768 * 768;
    const float* W1_l = W1 + (size_t)l * 768 * 3072;
    const float* W2_l = W2 + (size_t)l * 3072 * 768;

    // weight transposes (fp32 -> bf16 [N,K])
    k_wT<1><<<dim3(24, 24), blk, 0, stream>>>(Wq_l, wt_q, 768, 768);
    k_wT<1><<<dim3(24, 24), blk, 0, stream>>>(Wk_l, wt_k, 768, 768);
    k_wT<1><<<dim3(24, 24), blk, 0, stream>>>(Wv_l, wt_v, 768, 768);
    k_wT<0><<<dim3(24, 24), blk, 0, stream>>>(Wo_l, wt_o, 768, 768);
    k_wT<0><<<dim3(96, 24), blk, 0, stream>>>(W1_l, wt_1, 768, 3072);
    k_wT<0><<<dim3(24, 96), blk, 0, stream>>>(W2_l, wt_2, 3072, 768);

    k_ln<<<lnBlocks, blk, 0, stream>>>(x, 768, ln1_g + l * 768, ln1_b + l * 768,
                                       xn, MTOK);
    gemm_k<3><<<dim3(6, 50, 1), blk, 0, stream>>>(
        xn, 768, 0, wt_q, 768, 0, MTOK, 768, 768, nullptr, q, 0, 0,
        bq + l * 768, nullptr, 1.f);
    gemm_k<3><<<dim3(6, 50, 1), blk, 0, stream>>>(
        xn, 768, 0, wt_k, 768, 0, MTOK, 768, 768, nullptr, kb, 0, 0,
        bk + l * 768, nullptr, 1.f);
    gemm_k<4><<<dim3(6, 50, 1), blk, 0, stream>>>(
        xn, 768, 0, wt_v, 768, 0, MTOK, 768, 768, nullptr, vb, 0, 0,
        bv + l * 768, nullptr, 1.f);
    // scores = (Q @ K^T) / 8, batched over bh
    gemm_k<0><<<dim3(2, 2, 384), blk, 0, stream>>>(
        q, 64, (ull)197 * 64, kb, 64, (ull)197 * 64, 197, 197, 64, S, nullptr,
        208, (ull)208 * 208, nullptr, nullptr, 0.125f);
    k_softmax<<<(384 * 197 + 7) / 8, blk, 0, stream>>>(S, att);
    // O = att @ V -> [B,197,H*64]
    gemm_k<6><<<dim3(1, 2, 384), blk, 0, stream>>>(
        att, 208, (ull)208 * 208, vb, 208, (ull)64 * 208, 197, 64, 208, nullptr,
        ob, 0, 0, nullptr, nullptr, 1.f);
    // x += O @ Wo + bo
    gemm_k<1><<<dim3(6, 50, 1), blk, 0, stream>>>(
        ob, 768, 0, wt_o, 768, 0, MTOK, 768, 768, x, nullptr, 768, 0,
        bo + l * 768, nullptr, 1.f);
    // MLP
    k_ln<<<lnBlocks, blk, 0, stream>>>(x, 768, ln2_g + l * 768, ln2_b + l * 768,
                                       xn, MTOK);
    gemm_k<2><<<dim3(24, 50, 1), blk, 0, stream>>>(
        xn, 768, 0, wt_1, 768, 0, MTOK, 3072, 768, nullptr, hmid, 3072, 0,
        b1 + l * 3072, nullptr, 1.f);
    gemm_k<1><<<dim3(6, 50, 1), blk, 0, stream>>>(
        hmid, 3072, 0, wt_2, 3072, 0, MTOK, 768, 3072, x, nullptr, 768, 0,
        b2 + l * 768, nullptr, 1.f);
  }

  // ---- head ----
  k_ln<<<4, blk, 0, stream>>>(x, (long long)197 * 768, head_g, head_bn, xcls, B_);
  k_wT<0><<<dim3(32, 24), blk, 0, stream>>>(head_W, wt_h, 768, 1000);
  gemm_k<0><<<dim3(8, 1, 1), blk, 0, stream>>>(
      xcls, 768, 0, wt_h, 768, 0, B_, 1000, 768, (float*)d_out, nullptr, 1000,
      0, head_bias, nullptr, 1.f);
}
